// AM_70832600646102
// MI455X (gfx1250) — matrix operand order repaired and checked by reading, not compiled, not run on hardware
//
#include <hip/hip_runtime.h>
#include <hip/hip_bf16.h>
#include <math.h>

#define BB 512
#define NN 20
#define DD 128
#define HH 8
#define KDk 16
#define FFD 512
#define HIDD 512
#define LL 3

typedef __attribute__((ext_vector_type(16))) _Float16 v16h;
typedef __attribute__((ext_vector_type(8)))  float    v8f;

// ---------------------------------------------------------------------------
// WMMA GEMM, register-blocked: each wave computes a 16x64 output strip
// (4 accumulators), so one A fragment feeds 4 v_wmma_f32_16x16x32_f16.
//   C[M,N] = act( A[M,K](f16) * Bp[K,N](f16, fragment-packed) + bias + Cres )
// B is pre-swizzled into WMMA fragment order (see k_packB*): each lane's
// 16-half fragment is one contiguous 32-byte v16h load.
// C (f32) and Ch (f16 copy) are each optional.
// K % 32 == 0, M % 16 == 0, N % 64 == 0. A/B use contiguous k order, k = 16*hi + j (wave32).
// ---------------------------------------------------------------------------
__global__ void k_gemm(const _Float16* __restrict__ A, const _Float16* __restrict__ Bp,
                       const float* __restrict__ Cres, const float* __restrict__ bias,
                       float* __restrict__ C, _Float16* __restrict__ Ch,
                       int M, int N, int K, int act)
{
    int tilesN = N >> 6;               // 64-wide strips
    int tile   = blockIdx.x;
    int tm = tile / tilesN;
    int tn = tile - tm * tilesN;
    int lane = threadIdx.x;            // 0..31
    int hi   = lane >> 4;              // 0 or 1
    int l15  = lane & 15;

    int arow  = tm * 16 + l15;         // A: lane index = M row
    int ncol0 = tn * 64 + l15;         // first of 4 tile columns for this lane
    int crow0 = tm * 16 + hi * 8;      // C/D: VGPR r -> M = hi*8 + r
    int kblocks = K >> 5;

    const v16h* Bf = (const v16h*)Bp;  // fragment-packed: 32 v16h per 32x16 tile

    v8f acc[4];
#pragma unroll
    for (int t = 0; t < 4; ++t) {
        if (Cres) {
#pragma unroll
            for (int r = 0; r < 8; ++r)
                acc[t][r] = Cres[(size_t)(crow0 + r) * N + ncol0 + t * 16];
        } else {
#pragma unroll
            for (int r = 0; r < 8; ++r) acc[t][r] = 0.0f;
        }
        if (bias) {
            float bv = bias[ncol0 + t * 16];
#pragma unroll
            for (int r = 0; r < 8; ++r) acc[t][r] += bv;
        }
    }

    for (int kk = 0; kk < K; kk += 32) {
        // A fragment: half j -> k = kk + hi*16 + j
        v16h a;
        const _Float16* Ap = A + (size_t)arow * K + kk + hi * 16;
#pragma unroll
        for (int j = 0; j < 16; ++j)
            a[j] = Ap[j];

        int kb = kk >> 5;
#pragma unroll
        for (int t = 0; t < 4; ++t) {
            v16h b = Bf[((size_t)(tn * 4 + t) * kblocks + kb) * 32 + lane];
            acc[t] = __builtin_amdgcn_wmma_f32_16x16x32_f16(false, a, false, b,
                                                            (short)0, acc[t],
                                                            false, false);
        }
    }

#pragma unroll
    for (int t = 0; t < 4; ++t) {
#pragma unroll
        for (int r = 0; r < 8; ++r) {
            float v = acc[t][r];
            if (act == 1) v = v > 0.0f ? v : 0.0f;
            size_t idx = (size_t)(crow0 + r) * N + ncol0 + t * 16;
            if (C)  C[idx]  = v;
            if (Ch) Ch[idx] = (_Float16)v;
        }
    }
}

// ---------------------------------------------------------------------------
// Weight packing into WMMA B-fragment order.
// dst half index t: tile vi = t>>9, lane = (t>>4)&31, j = t&15
//   ntile = vi / (K/32), kb = vi % (K/32)
//   src element = B[kb*32 + (lane>>4)*16 + j, ntile*16 + (lane&15)]
// ---------------------------------------------------------------------------
__global__ void k_packBf(const float* __restrict__ src, _Float16* __restrict__ dst,
                         int K, int N)
{
    int t = blockIdx.x * blockDim.x + threadIdx.x;
    if (t >= K * N) return;
    int vi = t >> 9;  int wi = t & 511;
    int lane = wi >> 4, j = wi & 15;
    int kblocks = K >> 5;
    int ntile = vi / kblocks, kb = vi - ntile * kblocks;
    int k   = kb * 32 + ((lane >> 4) << 4) + j;
    int col = ntile * 16 + (lane & 15);
    dst[t] = (_Float16)src[(size_t)k * N + col];
}

__global__ void k_packBh(const _Float16* __restrict__ src, _Float16* __restrict__ dst,
                         int K, int N)
{
    int t = blockIdx.x * blockDim.x + threadIdx.x;
    if (t >= K * N) return;
    int vi = t >> 9;  int wi = t & 511;
    int lane = wi >> 4, j = wi & 15;
    int kblocks = K >> 5;
    int ntile = vi / kblocks, kb = vi - ntile * kblocks;
    int k   = kb * 32 + ((lane >> 4) << 4) + j;
    int col = ntile * 16 + (lane & 15);
    dst[t] = src[(size_t)k * N + col];
}

// fused QKV weight -> f16 row-major staging: dst[l][d][which*128+h*16+kd]
__global__ void k_pack_qkv(const float* __restrict__ Wq, const float* __restrict__ Wk,
                           const float* __restrict__ Wv, _Float16* __restrict__ dst)
{
    int t = blockIdx.x * blockDim.x + threadIdx.x;
    const int per = 3 * DD * DD;
    if (t >= LL * per) return;
    int l = t / per;  int r = t % per;
    int which = r / (DD * DD);  int e = r % (DD * DD);
    int d = e / DD;  int col = e % DD;
    int h = col / KDk;  int kd = col % KDk;
    const float* W = (which == 0) ? Wq : (which == 1) ? Wk : Wv;
    float v = W[((size_t)(l * HH + h) * DD + d) * KDk + kd];
    dst[(size_t)l * DD * (3 * DD) + (size_t)d * (3 * DD) + which * DD + col] = (_Float16)v;
}

__global__ void k_ctxq(const float* __restrict__ W_ph, const float* __restrict__ W_step,
                       float* __restrict__ ctxq)
{
    int d = blockIdx.x * blockDim.x + threadIdx.x;
    if (d >= DD) return;
    float s = 0.0f;
    for (int j = 0; j < 2 * DD; ++j) s += W_ph[j] * W_step[j * DD + d];
    ctxq[d] = s;
}

// ---------------------------------------------------------------------------
// Initial embedding + state (writes fp32 master + f16 GEMM operand copy)
// ---------------------------------------------------------------------------
__global__ void k_embed(const float* __restrict__ coords, const float* __restrict__ Wi,
                        const float* __restrict__ bi, float* __restrict__ x,
                        _Float16* __restrict__ xh)
{
    int t = blockIdx.x * blockDim.x + threadIdx.x;
    if (t >= BB * NN * DD) return;
    int d = t % DD;  int r = t / DD;   // r = b*NN + node
    float c0 = coords[2 * r], c1 = coords[2 * r + 1];
    float v = c0 * Wi[d] + c1 * Wi[DD + d] + bi[d];
    x[t]  = v;
    xh[t] = (_Float16)v;
}

__global__ void k_initstate(const float* __restrict__ coords, float* __restrict__ crd,
                            int* __restrict__ idx)
{
    int t = blockIdx.x * blockDim.x + threadIdx.x;
    if (t >= BB * NN) return;
    crd[2 * t]     = coords[2 * t];
    crd[2 * t + 1] = coords[2 * t + 1];
    idx[t] = t % NN;
}

// ---------------------------------------------------------------------------
// Encoder MHA core: softmax(QK^T/4)V per (b,h,row) from fused qkv buffer.
// qkv row layout: [0,128)=Q, [128,256)=K, [256,384)=V, head h at h*16.
// Output written directly as f16 (consumed only as GEMM A operand).
// ---------------------------------------------------------------------------
__global__ void k_attn(const float* __restrict__ qkv, _Float16* __restrict__ attH, int n)
{
    int t = blockIdx.x * blockDim.x + threadIdx.x;
    int total = BB * HH * n;
    if (t >= total) return;
    int b = t / (HH * n);  int r = t % (HH * n);
    int h = r / n;         int row = r % n;

    const float* qrow = qkv + (size_t)(b * n + row) * 384 + h * 16;
    float s[NN];
    float mx = -1e30f;
    for (int m = 0; m < n; ++m) {
        const float* krow = qkv + (size_t)(b * n + m) * 384 + 128 + h * 16;
        float d = 0.0f;
        for (int kd = 0; kd < 16; ++kd) d += qrow[kd] * krow[kd];
        d *= 0.25f;                       // 1/sqrt(16)
        s[m] = d;  if (d > mx) mx = d;
    }
    float sum = 0.0f;
    for (int m = 0; m < n; ++m) { s[m] = expf(s[m] - mx); sum += s[m]; }
    float inv = 1.0f / sum;
    float o[16];
    for (int kd = 0; kd < 16; ++kd) o[kd] = 0.0f;
    for (int m = 0; m < n; ++m) {
        const float* vrow = qkv + (size_t)(b * n + m) * 384 + 256 + h * 16;
        float w = s[m] * inv;
        for (int kd = 0; kd < 16; ++kd) o[kd] += w * vrow[kd];
    }
    _Float16* orow = attH + (size_t)(b * n + row) * DD + h * 16;
    for (int kd = 0; kd < 16; ++kd) orow[kd] = (_Float16)o[kd];
}

// ---------------------------------------------------------------------------
// BatchNorm (training stats over all rows, per-channel), in place on fp32,
// also emits f16 copy for downstream GEMM A operands. Block per channel.
// ---------------------------------------------------------------------------
__global__ void k_bn(float* __restrict__ y, _Float16* __restrict__ yh,
                     const float* __restrict__ g, const float* __restrict__ beta, int M)
{
    int c = blockIdx.x;                 // channel 0..127
    __shared__ float ss[256], sq[256];
    float s = 0.0f, q = 0.0f;
    for (int r = threadIdx.x; r < M; r += blockDim.x) {
        float v = y[(size_t)r * DD + c];
        s += v;  q += v * v;
    }
    ss[threadIdx.x] = s;  sq[threadIdx.x] = q;
    __syncthreads();
    for (int st = 128; st > 0; st >>= 1) {
        if ((int)threadIdx.x < st) { ss[threadIdx.x] += ss[threadIdx.x + st];
                                     sq[threadIdx.x] += sq[threadIdx.x + st]; }
        __syncthreads();
    }
    float mean = ss[0] / (float)M;
    float var  = sq[0] / (float)M - mean * mean;
    float sc = rsqrtf(var + 1e-5f) * g[c];
    float sh = beta[c] - mean * sc;
    for (int r = threadIdx.x; r < M; r += blockDim.x) {
        size_t i = (size_t)r * DD + c;
        float v = y[i] * sc + sh;
        y[i]  = v;
        yh[i] = (_Float16)v;
    }
}

// ---------------------------------------------------------------------------
// Decoder pieces
// ---------------------------------------------------------------------------
__global__ void k_gmean(const float* __restrict__ enc, float* __restrict__ gm,
                        _Float16* __restrict__ gmh, int n)
{
    int t = blockIdx.x * blockDim.x + threadIdx.x;
    if (t >= BB * DD) return;
    int b = t >> 7, d = t & 127;
    float s = 0.0f;
    for (int j = 0; j < n; ++j) s += enc[(size_t)(b * n + j) * DD + d];
    float v = s / (float)n;
    gm[t]  = v;
    gmh[t] = (_Float16)v;
}

// glimpse attention: per (b,h).  ewn cols [0,128)=gK, [128,256)=gV, [256,384)=lK
__global__ void k_glimpse(const float* __restrict__ q, const float* __restrict__ ewn,
                          _Float16* __restrict__ glH, int n)
{
    int t = blockIdx.x * blockDim.x + threadIdx.x;
    if (t >= BB * HH) return;
    int b = t >> 3, h = t & 7;
    const float* qh = q + (size_t)b * DD + h * 16;
    float s[NN];
    float mx = -1e30f;
    for (int j = 0; j < n; ++j) {
        const float* kr = ewn + (size_t)(b * n + j) * 384 + h * 16;
        float d = 0.0f;
        for (int kd = 0; kd < 16; ++kd) d += qh[kd] * kr[kd];
        d *= 0.25f;
        s[j] = d;  if (d > mx) mx = d;
    }
    float sum = 0.0f;
    for (int j = 0; j < n; ++j) { s[j] = expf(s[j] - mx); sum += s[j]; }
    float inv = 1.0f / sum;
    float o[16];
    for (int kd = 0; kd < 16; ++kd) o[kd] = 0.0f;
    for (int j = 0; j < n; ++j) {
        const float* vr = ewn + (size_t)(b * n + j) * 384 + 128 + h * 16;
        float w = s[j] * inv;
        for (int kd = 0; kd < 16; ++kd) o[kd] += w * vr[kd];
    }
    _Float16* gr = glH + (size_t)b * DD + h * 16;
    for (int kd = 0; kd < 16; ++kd) gr[kd] = (_Float16)o[kd];
}

__global__ void k_logits(const float* __restrict__ gl2, const float* __restrict__ ewn,
                         float* __restrict__ logits, int n)
{
    int t = blockIdx.x * blockDim.x + threadIdx.x;
    if (t >= BB * n) return;
    int b = t / n, j = t % n;
    const float* gr = gl2 + (size_t)b * DD;
    const float* lk = ewn + (size_t)(b * n + j) * 384 + 256;
    float d = 0.0f;
    for (int c = 0; c < DD; ++c) d += gr[c] * lk[c];
    logits[b * NN + j] = tanhf(d * 0.08838834764831845f) * 10.0f;  // /sqrt(128) * CLIP
}

__global__ void k_decode(const float* __restrict__ logits, const float* __restrict__ crd,
                         const int* __restrict__ nidx, float* __restrict__ prevc,
                         float* __restrict__ firstc, int* __restrict__ selb,
                         int* __restrict__ tours_i, float* __restrict__ out_logp,
                         float* __restrict__ out_irs, float* __restrict__ out_tours,
                         int n, int i)
{
    int b = blockIdx.x * blockDim.x + threadIdx.x;
    if (b >= BB) return;
    const float* lg = logits + b * NN;
    float mx = -1e30f;  int am = 0;
    for (int j = 0; j < n; ++j) if (lg[j] > mx) { mx = lg[j]; am = j; }
    float sum = 0.0f;
    for (int j = 0; j < n; ++j) sum += expf(lg[j] - mx);
    float lse = mx + logf(sum);
    out_logp[b * NN + i] = lg[am] - lse;

    float cx = crd[(size_t)(b * n + am) * 2];
    float cy = crd[(size_t)(b * n + am) * 2 + 1];
    if (i == 0) {
        firstc[2 * b] = cx;  firstc[2 * b + 1] = cy;
        out_irs[b * NN + i] = 0.0f;
    } else {
        float dx = cx - prevc[2 * b], dy = cy - prevc[2 * b + 1];
        out_irs[b * NN + i] = -sqrtf(dx * dx + dy * dy);
    }
    prevc[2 * b] = cx;  prevc[2 * b + 1] = cy;
    int node = nidx[b * n + am];
    tours_i[b * NN + i] = node;
    out_tours[b * NN + i] = (float)node;
    selb[b] = am;
}

// stable compaction dropping the selected node (double-buffered, fp32 + f16 copies)
__global__ void k_gather(const float* __restrict__ xs, float* __restrict__ xd,
                         const _Float16* __restrict__ xhs, _Float16* __restrict__ xhd,
                         const float* __restrict__ cs, float* __restrict__ cd,
                         const int* __restrict__ isrc, int* __restrict__ idst,
                         const int* __restrict__ selb, int n)
{
    int t = blockIdx.x * blockDim.x + threadIdx.x;
    int m = n - 1;
    if (t >= BB * m) return;
    int b = t / m, j = t % m;
    int s = selb[b];
    int src = j + (j >= s ? 1 : 0);
    const float*    xr  = xs  + (size_t)(b * n + src) * DD;
    float*          xw  = xd  + (size_t)(b * m + j) * DD;
    const _Float16* xhr = xhs + (size_t)(b * n + src) * DD;
    _Float16*       xhw = xhd + (size_t)(b * m + j) * DD;
    for (int d = 0; d < DD; ++d) { xw[d] = xr[d]; xhw[d] = xhr[d]; }
    cd[(size_t)(b * m + j) * 2]     = cs[(size_t)(b * n + src) * 2];
    cd[(size_t)(b * m + j) * 2 + 1] = cs[(size_t)(b * n + src) * 2 + 1];
    idst[b * m + j] = isrc[b * n + src];
}

__global__ void k_ccat(const float* __restrict__ gm, const float* __restrict__ W_ph,
                       _Float16* __restrict__ cc)
{
    int t = blockIdx.x * blockDim.x + threadIdx.x;
    if (t >= BB * 3 * DD) return;
    int b = t / (3 * DD), c = t % (3 * DD);
    cc[t] = (_Float16)((c < DD) ? gm[b * DD + c] : W_ph[c - DD]);
}

__global__ void k_vals(const float* __restrict__ h1, const float* __restrict__ Wc2,
                       const float* __restrict__ bc2, float* __restrict__ out, int i)
{
    int b = blockIdx.x * blockDim.x + threadIdx.x;
    if (b >= BB) return;
    float s = bc2[0];
    for (int j = 0; j < HIDD; ++j) s += h1[(size_t)b * HIDD + j] * Wc2[j];
    out[b * NN + i] = s;
}

__global__ void k_final(const float* __restrict__ coords, const int* __restrict__ tours,
                        const float* __restrict__ firstc, const float* __restrict__ prevc,
                        float* __restrict__ cost_out, float* __restrict__ rf_out)
{
    int b = blockIdx.x * blockDim.x + threadIdx.x;
    if (b >= BB) return;
    float c = 0.0f;
    for (int t = 0; t < NN - 1; ++t) {
        int a0 = tours[b * NN + t], a1 = tours[b * NN + t + 1];
        float dx = coords[(size_t)(b * NN + a1) * 2]     - coords[(size_t)(b * NN + a0) * 2];
        float dy = coords[(size_t)(b * NN + a1) * 2 + 1] - coords[(size_t)(b * NN + a0) * 2 + 1];
        c += sqrtf(dx * dx + dy * dy);
    }
    int a0 = tours[b * NN], a1 = tours[b * NN + NN - 1];
    float dx = coords[(size_t)(b * NN + a0) * 2]     - coords[(size_t)(b * NN + a1) * 2];
    float dy = coords[(size_t)(b * NN + a0) * 2 + 1] - coords[(size_t)(b * NN + a1) * 2 + 1];
    cost_out[b] = c + sqrtf(dx * dx + dy * dy);
    float fx = firstc[2 * b] - prevc[2 * b], fy = firstc[2 * b + 1] - prevc[2 * b + 1];
    rf_out[b] = -sqrtf(fx * fx + fy * fy);
}

// ---------------------------------------------------------------------------
static inline int cdiv(int a, int b) { return (a + b - 1) / b; }

extern "C" void kernel_launch(void* const* d_in, const int* in_sizes, int n_in,
                              void* d_out, int out_size, void* d_ws, size_t ws_size,
                              hipStream_t stream)
{
    (void)in_sizes; (void)n_in; (void)out_size; (void)ws_size;
    const float* coords = (const float*)d_in[0];
    const float* Wi     = (const float*)d_in[1];
    const float* bi     = (const float*)d_in[2];
    const float* W_ph   = (const float*)d_in[3];
    const float* Wq     = (const float*)d_in[4];
    const float* Wk     = (const float*)d_in[5];
    const float* Wv     = (const float*)d_in[6];
    const float* Wo     = (const float*)d_in[7];
    const float* g1     = (const float*)d_in[8];
    const float* b1     = (const float*)d_in[9];
    const float* fW1    = (const float*)d_in[10];
    const float* fb1    = (const float*)d_in[11];
    const float* fW2    = (const float*)d_in[12];
    const float* fb2    = (const float*)d_in[13];
    const float* g2     = (const float*)d_in[14];
    const float* b2     = (const float*)d_in[15];
    const float* W_node = (const float*)d_in[16];
    const float* W_fixed= (const float*)d_in[17];
    const float* W_step = (const float*)d_in[18];
    const float* W_out  = (const float*)d_in[19];
    const float* Wc1    = (const float*)d_in[20];
    const float* bc1    = (const float*)d_in[21];
    const float* Wc2    = (const float*)d_in[22];
    const float* bc2    = (const float*)d_in[23];

    // ---- workspace carve ----
    char* ws = (char*)d_ws;  size_t off = 0;
    auto alloc = [&](size_t bytes) -> void* {
        void* p = ws + off;
        off = (off + bytes + 255) & ~(size_t)255;
        return p;
    };
    const int BNmax = BB * NN;
    // fp32 masters
    float* enc   = (float*)alloc((size_t)BNmax * DD * 4);
    float* tmp   = (float*)alloc((size_t)BNmax * DD * 4);
    float* qkv   = (float*)alloc((size_t)BNmax * 3 * DD * 4);
    float* ewn   = (float*)alloc((size_t)BNmax * 3 * DD * 4);
    float* xA    = (float*)alloc((size_t)BNmax * DD * 4);
    float* xB    = (float*)alloc((size_t)BNmax * DD * 4);
    float* crdA  = (float*)alloc((size_t)BNmax * 2 * 4);
    float* crdB  = (float*)alloc((size_t)BNmax * 2 * 4);
    int*   idxA  = (int*)  alloc((size_t)BNmax * 4);
    int*   idxB  = (int*)  alloc((size_t)BNmax * 4);
    float* gmean = (float*)alloc((size_t)BB * DD * 4);
    float* qbuf  = (float*)alloc((size_t)BB * DD * 4);
    float* glim2 = (float*)alloc((size_t)BB * DD * 4);
    float* logits= (float*)alloc((size_t)BB * NN * 4);
    float* h1    = (float*)alloc((size_t)BB * HIDD * 4);
    float* ctxq  = (float*)alloc((size_t)DD * 4);
    float* prevc = (float*)alloc((size_t)BB * 2 * 4);
    float* firstc= (float*)alloc((size_t)BB * 2 * 4);
    int*   selb  = (int*)  alloc((size_t)BB * 4);
    int*   tours_i=(int*)  alloc((size_t)BB * NN * 4);
    // f16 GEMM operands
    _Float16* xhA  = (_Float16*)alloc((size_t)BNmax * DD * 2);
    _Float16* xhB  = (_Float16*)alloc((size_t)BNmax * DD * 2);
    _Float16* encH = (_Float16*)alloc((size_t)BNmax * DD * 2);
    _Float16* tmpH = (_Float16*)alloc((size_t)BNmax * DD * 2);
    _Float16* attH = (_Float16*)alloc((size_t)BNmax * DD * 2);
    _Float16* f1h  = (_Float16*)alloc((size_t)BNmax * FFD * 2);
    _Float16* gmH  = (_Float16*)alloc((size_t)BB * DD * 2);
    _Float16* glH  = (_Float16*)alloc((size_t)BB * DD * 2);
    _Float16* ccH  = (_Float16*)alloc((size_t)BB * 3 * DD * 2);
    // f16 packed weights (WMMA B-fragment order)
    _Float16* WqkvRM= (_Float16*)alloc((size_t)LL * DD * 3 * DD * 2);  // staging
    _Float16* WqkvP = (_Float16*)alloc((size_t)LL * DD * 3 * DD * 2);
    _Float16* WoP   = (_Float16*)alloc((size_t)LL * DD * DD * 2);
    _Float16* W1P   = (_Float16*)alloc((size_t)LL * DD * FFD * 2);
    _Float16* W2P   = (_Float16*)alloc((size_t)LL * FFD * DD * 2);
    _Float16* WnodeP= (_Float16*)alloc((size_t)DD * 3 * DD * 2);
    _Float16* WfixP = (_Float16*)alloc((size_t)DD * DD * 2);
    _Float16* WoutP = (_Float16*)alloc((size_t)DD * DD * 2);
    _Float16* Wc1P  = (_Float16*)alloc((size_t)3 * DD * HIDD * 2);

    // ---- output layout: log_ps, irs, vals, cost, reward_final, tours ----
    float* o = (float*)d_out;
    float* out_logp  = o;
    float* out_irs   = o + (size_t)BB * NN;
    float* out_vals  = o + (size_t)2 * BB * NN;
    float* out_cost  = o + (size_t)3 * BB * NN;
    float* out_rf    = out_cost + BB;
    float* out_tours = out_rf + BB;

    // ---- pack weights to WMMA fragment order ----
    {
        int n = LL * 3 * DD * DD;
        k_pack_qkv<<<cdiv(n, 256), 256, 0, stream>>>(Wq, Wk, Wv, WqkvRM);
        for (int l = 0; l < LL; ++l) {
            n = DD * 3 * DD;
            k_packBh<<<cdiv(n, 256), 256, 0, stream>>>(
                WqkvRM + (size_t)l * n, WqkvP + (size_t)l * n, DD, 3 * DD);
            n = DD * DD;
            k_packBf<<<cdiv(n, 256), 256, 0, stream>>>(
                Wo + (size_t)l * n, WoP + (size_t)l * n, DD, DD);
            n = DD * FFD;
            k_packBf<<<cdiv(n, 256), 256, 0, stream>>>(
                fW1 + (size_t)l * n, W1P + (size_t)l * n, DD, FFD);
            n = FFD * DD;
            k_packBf<<<cdiv(n, 256), 256, 0, stream>>>(
                fW2 + (size_t)l * n, W2P + (size_t)l * n, FFD, DD);
        }
        n = DD * 3 * DD;   k_packBf<<<cdiv(n, 256), 256, 0, stream>>>(W_node, WnodeP, DD, 3 * DD);
        n = DD * DD;       k_packBf<<<cdiv(n, 256), 256, 0, stream>>>(W_fixed, WfixP, DD, DD);
        n = DD * DD;       k_packBf<<<cdiv(n, 256), 256, 0, stream>>>(W_out, WoutP, DD, DD);
        n = 3 * DD * HIDD; k_packBf<<<cdiv(n, 256), 256, 0, stream>>>(Wc1, Wc1P, 3 * DD, HIDD);
        k_ctxq<<<1, 128, 0, stream>>>(W_ph, W_step, ctxq);
    }

    // ---- initial embeddings + decode state ----
    k_embed<<<cdiv(BB * NN * DD, 256), 256, 0, stream>>>(coords, Wi, bi, xA, xhA);
    k_initstate<<<cdiv(BB * NN, 256), 256, 0, stream>>>(coords, crdA, idxA);

    float*    xcur  = xA;   float*    xnext  = xB;
    _Float16* xhcur = xhA;  _Float16* xhnext = xhB;
    float*    ccur  = crdA; float*    cnext  = crdB;
    int*      icur  = idxA; int*      inext  = idxB;

    for (int i = 0; i < NN; ++i) {
        int n = NN - i;
        int M = BB * n;

        // ---------------- encoder (3 layers) ----------------
        for (int l = 0; l < LL; ++l) {
            const float*    A0f = (l == 0) ? xcur  : enc;
            const _Float16* A0h = (l == 0) ? xhcur : encH;
            // fused QKV projection
            k_gemm<<<(M / 16) * ((3 * DD) / 64), 32, 0, stream>>>(
                A0h, WqkvP + (size_t)l * DD * 3 * DD, nullptr, nullptr,
                qkv, nullptr, M, 3 * DD, DD, 0);
            k_attn<<<cdiv(BB * HH * n, 256), 256, 0, stream>>>(qkv, attH, n);
            // out-projection + residual (Cres folded into accumulator init)
            k_gemm<<<(M / 16) * (DD / 64), 32, 0, stream>>>(
                attH, WoP + (size_t)l * DD * DD, A0f, nullptr,
                tmp, nullptr, M, DD, DD, 0);
            k_bn<<<DD, 256, 0, stream>>>(tmp, tmpH, g1 + l * DD, b1 + l * DD, M);
            // FF1 (relu, f16-only output: consumed only as next GEMM A)
            k_gemm<<<(M / 16) * (FFD / 64), 32, 0, stream>>>(
                tmpH, W1P + (size_t)l * DD * FFD, nullptr, fb1 + l * FFD,
                nullptr, f1h, M, FFD, DD, 1);
            // FF2 + residual
            k_gemm<<<(M / 16) * (DD / 64), 32, 0, stream>>>(
                f1h, W2P + (size_t)l * FFD * DD, tmp, fb2 + l * DD,
                enc, nullptr, M, DD, FFD, 0);
            k_bn<<<DD, 256, 0, stream>>>(enc, encH, g2 + l * DD, b2 + l * DD, M);
        }

        // ---------------- decoder ----------------
        k_gmean<<<cdiv(BB * DD, 256), 256, 0, stream>>>(enc, gmean, gmH, n);
        k_gemm<<<(BB / 16) * (DD / 64), 32, 0, stream>>>(
            gmH, WfixP, nullptr, ctxq, qbuf, nullptr, BB, DD, DD, 0);
        k_gemm<<<(M / 16) * ((3 * DD) / 64), 32, 0, stream>>>(
            encH, WnodeP, nullptr, nullptr, ewn, nullptr, M, 3 * DD, DD, 0);
        k_glimpse<<<cdiv(BB * HH, 256), 256, 0, stream>>>(qbuf, ewn, glH, n);
        k_gemm<<<(BB / 16) * (DD / 64), 32, 0, stream>>>(
            glH, WoutP, nullptr, nullptr, glim2, nullptr, BB, DD, DD, 0);
        k_logits<<<cdiv(BB * n, 256), 256, 0, stream>>>(glim2, ewn, logits, n);
        k_decode<<<cdiv(BB, 256), 256, 0, stream>>>(
            logits, ccur, icur, prevc, firstc, selb, tours_i,
            out_logp, out_irs, out_tours, n, i);

        // critic value head
        k_ccat<<<cdiv(BB * 3 * DD, 256), 256, 0, stream>>>(gmean, W_ph, ccH);
        k_gemm<<<(BB / 16) * (HIDD / 64), 32, 0, stream>>>(
            ccH, Wc1P, nullptr, bc1, h1, nullptr, BB, HIDD, 3 * DD, 1);
        k_vals<<<cdiv(BB, 256), 256, 0, stream>>>(h1, Wc2, bc2, out_vals, i);

        // compact state (double-buffered to avoid in-place race)
        if (i < NN - 1) {
            k_gather<<<cdiv(BB * (n - 1), 256), 256, 0, stream>>>(
                xcur, xnext, xhcur, xhnext, ccur, cnext, icur, inext, selb, n);
            float* tf;  int* ti;  _Float16* th;
            tf = xcur;  xcur  = xnext;  xnext  = tf;
            th = xhcur; xhcur = xhnext; xhnext = th;
            tf = ccur;  ccur  = cnext;  cnext  = tf;
            ti = icur;  icur  = inext;  inext  = ti;
        }
    }

    k_final<<<cdiv(BB, 256), 256, 0, stream>>>(coords, tours_i, firstc, prevc,
                                               out_cost, out_rf);
}
